// GraphCVAE_67680094650554
// MI455X (gfx1250) — matrix operand order repaired and checked by reading, not compiled, not run on hardware
//
#include <hip/hip_runtime.h>
#include <cstdint>
#include <climits>

// ---------------- problem constants (match reference) ----------------
#define NN   40960     // nodes
#define EE   245760    // edges (before self loops)
#define BB   512       // graphs
#define FDIM 128
#define NH   8
#define HID  1024      // NH*FDIM
#define GWID 3200      // 128 + 3*1024
#define CDIV(a,b) (((a)+(b)-1)/(b))

// output layout (floats): pos[N,2] size[N,2] theta[N,1] mu[B,128] logvar[B,128]
#define OUT_POS   0
#define OUT_SIZE  (NN*2)
#define OUT_THETA (NN*4)
#define OUT_MU    (NN*5)
#define OUT_LV    (NN*5 + BB*128)

// ---------------- WMMA bf16 GEMM ----------------
typedef __attribute__((ext_vector_type(16))) __bf16 v16bf;
typedef __attribute__((ext_vector_type(8)))  float  v8f;

__device__ __forceinline__ unsigned short f2bf(float f) {
  unsigned int u = __float_as_uint(f);
  unsigned int r = ((u >> 16) & 1u) + 0x7FFFu;   // round-to-nearest-even
  return (unsigned short)((u + r) >> 16);
}
__device__ __forceinline__ unsigned int f2bf2(float lo, float hi) {
  return (unsigned int)f2bf(lo) | ((unsigned int)f2bf(hi) << 16);
}

union FragU { unsigned int u[8]; v16bf v; };

#define BM 128
#define BN 128
#define BK 64

// C[M,Ncols] = A[M,K] @ Bw[K,Ncols]  (+bias per column) (+relu)
// M % 128 == 0, Ncols % 128 == 0, K % 64 == 0 for all call sites.
// 256 threads = 8 waves in a 4(m) x 2(n) grid; each wave owns 32x64 (2x4 WMMA tiles).
// BK=64 -> two 32-K WMMA slices per stage: 16 WMMAs per 2 barriers.
__global__ __launch_bounds__(256) void k_gemm_bf16(
    const float* __restrict__ A, const float* __restrict__ Bw,
    float* __restrict__ C, const float* __restrict__ bias,
    int M, int Ncols, int K, int do_relu)
{
  __shared__ unsigned int As[BM][BK/2];   // bf16 pairs, row-major [m][k]   16KB
  __shared__ unsigned int Bs[BN][BK/2];   // bf16 pairs, col-major [n][k]   16KB
  unsigned short* BsS = (unsigned short*)Bs;

  const int tid  = threadIdx.x;
  const int wave = tid >> 5;
  const int lane = tid & 31;
  const int wm   = wave >> 1;        // 0..3
  const int wn   = wave & 1;         // 0..1
  const int lr   = lane & 15;
  const int hi   = lane >> 4;        // 0/1 (lane half)
  const int bm0  = blockIdx.x * BM;
  const int bn0  = blockIdx.y * BN;

  v8f acc[2][4];
  #pragma unroll
  for (int a = 0; a < 2; ++a)
    #pragma unroll
    for (int b = 0; b < 4; ++b)
      #pragma unroll
      for (int e = 0; e < 8; ++e) acc[a][b][e] = 0.f;

  for (int k0 = 0; k0 < K; k0 += BK) {
    // stage A tile 128x64 (8192 floats, 8 x float4 per thread)
    #pragma unroll
    for (int i = 0; i < 8; ++i) {
      int e  = (i * 256 + tid) * 4;
      int r  = e >> 6, c = e & 63;                       // c % 4 == 0
      float4 f = *(const float4*)&A[(size_t)(bm0 + r) * K + k0 + c];
      uint2 dv;
      dv.x = f2bf2(f.x, f.y);
      dv.y = f2bf2(f.z, f.w);
      *(uint2*)&As[r][c >> 1] = dv;
    }
    // stage B tile 64x128 -> transposed [n][k] (b16 scatter stores)
    #pragma unroll
    for (int i = 0; i < 8; ++i) {
      int e  = (i * 256 + tid) * 4;
      int kk = e >> 7, nn = e & 127;                     // nn % 4 == 0
      float4 f = *(const float4*)&Bw[(size_t)(k0 + kk) * Ncols + bn0 + nn];
      BsS[(nn + 0) * BK + kk] = f2bf(f.x);
      BsS[(nn + 1) * BK + kk] = f2bf(f.y);
      BsS[(nn + 2) * BK + kk] = f2bf(f.z);
      BsS[(nn + 3) * BK + kk] = f2bf(f.w);
    }
    __syncthreads();

    #pragma unroll
    for (int ks = 0; ks < 2; ++ks) {                     // two 32-K WMMA slices
      FragU af[2], bfr[4];
      // A frag (16x32 bf16): lane row = lr; dwords 0-3 -> K{0..7}+8*hi, 4-7 -> K{16..23}+8*hi
      #pragma unroll
      for (int mt = 0; mt < 2; ++mt) {
        int row = wm * 32 + mt * 16 + lr;
        const unsigned int* Ar = &As[row][ks * 16];
        #pragma unroll
        for (int j = 0; j < 8; ++j) {
          int di = ((j >= 4) ? 8 : 0) + (hi ? 4 : 0) + (j & 3);
          af[mt].u[j] = Ar[di];
        }
      }
      // B frag (32x16 bf16): lane col = lr; dwords 0-3 -> K{0..7}+8*hi, 4-7 -> K{16..23}+8*hi
      #pragma unroll
      for (int nt = 0; nt < 4; ++nt) {
        int col = wn * 64 + nt * 16 + lr;
        const unsigned int* Br = &Bs[col][ks * 16];
        #pragma unroll
        for (int j = 0; j < 8; ++j) bfr[nt].u[j] = Br[((j >= 4) ? 8 : 0) + (hi ? 4 : 0) + (j & 3)];
      }
      #pragma unroll
      for (int mt = 0; mt < 2; ++mt)
        #pragma unroll
        for (int nt = 0; nt < 4; ++nt)
          acc[mt][nt] = __builtin_amdgcn_wmma_f32_16x16x32_bf16(
              false, af[mt].v, false, bfr[nt].v, (short)0, acc[mt][nt], false, false);
    }
    __syncthreads();
  }

  // D layout: VGPR r -> M = r (+8 for lane half 1), N = lane&15
  const int rowadd = hi ? 8 : 0;
  #pragma unroll
  for (int nt = 0; nt < 4; ++nt) {
    int n = bn0 + wn * 64 + nt * 16 + lr;
    float bv = bias ? bias[n] : 0.f;                     // one load per column tile
    #pragma unroll
    for (int mt = 0; mt < 2; ++mt) {
      #pragma unroll
      for (int r = 0; r < 8; ++r) {
        int m = bm0 + wm * 32 + mt * 16 + rowadd + r;
        float v = acc[mt][nt][r] + bv;
        v = do_relu ? fmaxf(v, 0.f) : v;
        C[(size_t)m * Ncols + n] = v;
      }
    }
  }
}

// ---------------- small utility kernels ----------------
__global__ void k_fill_int(int* p, int val, int n) {
  int i = blockIdx.x * 256 + threadIdx.x;
  if (i < n) p[i] = val;
}

__global__ void k_count_dst(const int* __restrict__ dst, int e, int* counts) {
  int i = blockIdx.x * 256 + threadIdx.x;
  if (i < e) atomicAdd(&counts[dst[i]], 1);
}

__global__ void k_scan_block(const int* __restrict__ in, int* out, int* bsums, int n) {
  __shared__ int sh[256];
  int gid = blockIdx.x * 256 + threadIdx.x;
  int v = (gid < n) ? in[gid] : 0;
  sh[threadIdx.x] = v;
  __syncthreads();
  for (int ofs = 1; ofs < 256; ofs <<= 1) {
    int t = (threadIdx.x >= ofs) ? sh[threadIdx.x - ofs] : 0;
    __syncthreads();
    sh[threadIdx.x] += t;
    __syncthreads();
  }
  if (gid < n) out[gid] = sh[threadIdx.x] - v;   // exclusive
  if (threadIdx.x == 255) bsums[blockIdx.x] = sh[255];
}

__global__ void k_scan_sums(int* bsums, int nb) {
  if (blockIdx.x == 0 && threadIdx.x == 0) {
    int acc = 0;
    for (int i = 0; i < nb; ++i) { int t = bsums[i]; bsums[i] = acc; acc += t; }
  }
}

__global__ void k_scan_add(int* out, const int* bsums, int n) {
  int gid = blockIdx.x * 256 + threadIdx.x;
  if (gid < n) out[gid] += bsums[blockIdx.x];
}

__global__ void k_scatter_edges(const int* __restrict__ src, const int* __restrict__ dst,
                                int e, const int* __restrict__ starts, int* cursor, int* slist) {
  int i = blockIdx.x * 256 + threadIdx.x;
  if (i < e) {
    int d = dst[i];
    int p = atomicAdd(&cursor[d], 1);
    slist[starts[d] + p] = src[i];
  }
}

__global__ void k_scatter_self(int n, const int* __restrict__ starts, int* cursor, int* slist) {
  int i = blockIdx.x * 256 + threadIdx.x;
  if (i < n) {
    int p = atomicAdd(&cursor[i], 1);
    slist[starts[i] + p] = i;
  }
}

// canonical neighbor order -> deterministic fp sums
__global__ void k_sort_lists(const int* __restrict__ starts, const int* __restrict__ counts,
                             int* slist, int n) {
  int node = blockIdx.x * 256 + threadIdx.x;
  if (node >= n) return;
  int s = starts[node], c = counts[node];
  for (int i = 1; i < c; ++i) {
    int key = slist[s + i]; int j = i - 1;
    while (j >= 0 && slist[s + j] > key) { slist[s + j + 1] = slist[s + j]; --j; }
    slist[s + j + 1] = key;
  }
}

__global__ void k_graph_ranges(const int* __restrict__ batch, int n, int* gstart, int* gcount) {
  int i = blockIdx.x * 256 + threadIdx.x;
  if (i < n) {
    int b = batch[i];
    atomicMin(&gstart[b], i);
    atomicAdd(&gcount[b], 1);
  }
}

__global__ void k_order(const int* __restrict__ batch, const int* __restrict__ gstart,
                        int* order, int n) {
  int i = blockIdx.x * 256 + threadIdx.x;
  if (i < n) order[i] = i - gstart[batch[i]];
}

// ---------------- model-specific kernels ----------------
// cat[N,256]: [0:128)=relu(nf@bboxW+b), [128:256)=relu(mask_embed[mask])
__global__ void k_enc_cat(const float* __restrict__ nf5, const float* __restrict__ W,
                          const float* __restrict__ b, const float* __restrict__ membed,
                          const int* __restrict__ mask, float* __restrict__ cat) {
  int i = blockIdx.x * 256 + threadIdx.x;
  if (i >= NN * 256) return;
  int n = i >> 8, c = i & 255;
  float v;
  if (c < 128) {
    v = b[c];
    #pragma unroll
    for (int k = 0; k < 5; ++k) v += nf5[n * 5 + k] * W[k * 128 + c];
  } else {
    v = membed[mask[n] * 128 + (c - 128)];
  }
  cat[(size_t)n * 256 + c] = v > 0.f ? v : 0.f;
}

__global__ void k_lin5_relu(const float* __restrict__ in5, const float* __restrict__ W,
                            const float* __restrict__ b, float* __restrict__ out) {
  int i = blockIdx.x * 256 + threadIdx.x;
  if (i >= NN * 128) return;
  int n = i >> 7, c = i & 127;
  float v = b[c];
  #pragma unroll
  for (int k = 0; k < 5; ++k) v += in5[n * 5 + k] * W[k * 128 + c];
  out[i] = v > 0.f ? v : 0.f;
}

// a_src[n,h], a_dst[n,h] from xw [N,8,128]
__global__ void k_att(const float* __restrict__ xw, const float* __restrict__ att_src,
                      const float* __restrict__ att_dst, float* __restrict__ a_src,
                      float* __restrict__ a_dst) {
  int i = blockIdx.x * 256 + threadIdx.x;
  if (i >= NN * NH) return;
  int n = i >> 3, h = i & 7;
  const float* row = xw + (size_t)n * HID + h * 128;
  float s0 = 0.f, s1 = 0.f;
  for (int c = 0; c < 128; ++c) {
    float v = row[c];
    s0 += v * att_src[h * 128 + c];
    s1 += v * att_dst[h * 128 + c];
  }
  a_src[i] = s0;
  a_dst[i] = s1;
}

// one block per dst node: softmax over incident edges + weighted sum of xw rows
__global__ __launch_bounds__(256) void k_gat_aggregate(
    const float* __restrict__ xw, const float* __restrict__ a_src,
    const float* __restrict__ a_dst, const int* __restrict__ starts,
    const int* __restrict__ counts, const int* __restrict__ slist,
    const float* __restrict__ bias, float* __restrict__ out)
{
  int n = blockIdx.x;
  int t = threadIdx.x;
  __shared__ float s_max[NH], s_den[NH], s_wv[NH];
  __shared__ int s_src;

  int st = starts[n], cnt = counts[n];   // >=1 (self loop)

  if (t < NH) {                          // pass A: per-head max of leaky_relu
    float ad = a_dst[n * NH + t];
    float m = -INFINITY;
    for (int i = 0; i < cnt; ++i) {
      int s = slist[st + i];
      float v = a_src[s * NH + t] + ad;
      v = v > 0.f ? v : 0.2f * v;
      m = fmaxf(m, v);
    }
    s_max[t] = m;
    s_den[t] = 0.f;
  }
  __syncthreads();

  float accv[4] = {0.f, 0.f, 0.f, 0.f};
  for (int i = 0; i < cnt; ++i) {        // pass B: numerator + denominator
    if (t < NH) {
      int s = slist[st + i];
      if (t == 0) s_src = s;
      float v = a_src[s * NH + t] + a_dst[n * NH + t];
      v = v > 0.f ? v : 0.2f * v;
      float w = __expf(v - s_max[t]);
      s_wv[t] = w;
      s_den[t] += w;
    }
    __syncthreads();
    const float* row = xw + (size_t)s_src * HID;
    #pragma unroll
    for (int j = 0; j < 4; ++j) {
      int c = t * 4 + j;
      accv[j] += s_wv[c >> 7] * row[c];
    }
    __syncthreads();
  }

  #pragma unroll
  for (int j = 0; j < 4; ++j) {
    int c = t * 4 + j;
    float v = accv[j] / s_den[c >> 7] + bias[c];
    out[(size_t)n * HID + c] = v > 0.f ? v : 0.f;   // relu(gat_conv(...))
  }
}

// global_max_pool into concatenated g buffer at column offset
__global__ void k_pool(const float* __restrict__ x, int C, const int* __restrict__ gstart,
                       const int* __restrict__ gcount, float* __restrict__ g, int colofs) {
  int b = blockIdx.x;
  int c = blockIdx.y * 256 + threadIdx.x;
  if (c >= C) return;
  int s = gstart[b], cnt = gcount[b];
  float m = -INFINITY;
  for (int i = 0; i < cnt; ++i) m = fmaxf(m, x[(size_t)(s + i) * C + c]);
  g[(size_t)b * GWID + colofs + c] = m;
}

__global__ void k_reparam(const float* __restrict__ mu, const float* __restrict__ lv,
                          const float* __restrict__ eps, float* __restrict__ z) {
  int i = blockIdx.x * 256 + threadIdx.x;
  if (i < BB * 128) z[i] = __expf(0.5f * lv[i]) * eps[i] + mu[i];
}

__global__ void k_concat2(const float* __restrict__ a, const float* __restrict__ b,
                          float* __restrict__ o) {
  int i = blockIdx.x * 256 + threadIdx.x;
  if (i >= BB * 256) return;
  int r = i >> 8, c = i & 255;
  o[i] = (c < 128) ? a[r * 128 + c] : b[r * 128 + (c - 128)];
}

__global__ void k_dec_input(const float* __restrict__ zc, const int* __restrict__ batch,
                            float* __restrict__ x) {
  int i = blockIdx.x * 256 + threadIdx.x;
  if (i >= NN * 128) return;
  int n = i >> 7, c = i & 127;
  float v = zc[batch[n] * 128 + c];
  x[i] = v > 0.f ? v : 0.f;
}

// xw[n,:] += W[128 + order[n], :]  (one-hot position columns of dec gat0)
__global__ void k_add_onehot(float* __restrict__ xw, const float* __restrict__ W,
                             const int* __restrict__ order) {
  int n = blockIdx.x;
  const float* row = W + (size_t)(128 + order[n]) * HID;
  for (int j = threadIdx.x; j < HID; j += 256)
    xw[(size_t)n * HID + j] += row[j];
}

// out[n, 0..oc) = t128[n,:] @ W[128,oc] + b
__global__ void k_head_out(const float* __restrict__ t128, const float* __restrict__ W,
                           const float* __restrict__ b, float* __restrict__ out, int oc) {
  int i = blockIdx.x * 256 + threadIdx.x;
  if (i >= NN * oc) return;
  int n = i / oc, c = i % oc;
  const float* r = t128 + (size_t)n * 128;
  float v = b[c];
  for (int k = 0; k < 128; ++k) v += r[k] * W[k * oc + c];
  out[i] = v;
}

// ---------------- host orchestration ----------------
struct GatP { const float *W, *asrc, *adst, *b; };

extern "C" void kernel_launch(void* const* d_in, const int* in_sizes, int n_in,
                              void* d_out, int out_size, void* d_ws, size_t ws_size,
                              hipStream_t stream) {
  (void)in_sizes; (void)n_in; (void)out_size; (void)ws_size;

  // ---- inputs (setup_inputs dict order; params flattened in insertion order) ----
  const float* node_features = (const float*)d_in[0];
  const float* cond_street   = (const float*)d_in[1];
  const float* eps           = (const float*)d_in[2];
  int p = 3;
  auto nx = [&]() { return (const float*)d_in[p++]; };
  const float* enc_bbox_W = nx(); const float* enc_bbox_b = nx();
  const float* enc_mask_embed = nx();
  const float* enc_node_W = nx(); const float* enc_node_b = nx();
  GatP enc_g[3];
  for (int j = 0; j < 3; ++j) { enc_g[j].W = nx(); enc_g[j].asrc = nx(); enc_g[j].adst = nx(); enc_g[j].b = nx(); }
  const float* enc_agg_W = nx(); const float* enc_agg_b = nx();
  const float* mu_W = nx();  const float* mu_b = nx();
  const float* var_W = nx(); const float* var_b = nx();
  const float* cond_bbox_W = nx(); const float* cond_bbox_b = nx();
  GatP cond_g[3];
  for (int j = 0; j < 3; ++j) { cond_g[j].W = nx(); cond_g[j].asrc = nx(); cond_g[j].adst = nx(); cond_g[j].b = nx(); }
  const float* cond_agg_W = nx(); const float* cond_agg_b = nx();
  const float* init_W = nx(); const float* init_b = nx();
  GatP dec_g[3];
  for (int j = 0; j < 3; ++j) { dec_g[j].W = nx(); dec_g[j].asrc = nx(); dec_g[j].adst = nx(); dec_g[j].b = nx(); }
  const float* pos_W = nx();   const float* pos_b = nx();
  const float* fc_pos_W = nx(); const float* fc_pos_b = nx();
  const float* size_W = nx();  const float* size_b = nx();
  const float* fc_size_W = nx(); const float* fc_size_b = nx();
  const float* theta_W = nx(); const float* theta_b = nx();
  const float* fc_theta_W = nx(); const float* fc_theta_b = nx();
  const int* building_mask = (const int*)d_in[p++];
  const int* edge_index    = (const int*)d_in[p++];
  const int* batch         = (const int*)d_in[p++];

  float* out = (float*)d_out;

  // ---- workspace bump allocator ----
  char* w = (char*)d_ws;
  auto alloc = [&](size_t bytes) -> void* {
    void* q = (void*)w;
    w += (bytes + 255) & ~(size_t)255;
    return q;
  };
  float* xbuf   = (float*)alloc((size_t)NN * HID * 4);
  float* xwbuf  = (float*)alloc((size_t)NN * HID * 4);
  float* a_src  = (float*)alloc((size_t)NN * NH * 4);
  float* a_dst  = (float*)alloc((size_t)NN * NH * 4);
  float* gbuf   = (float*)alloc((size_t)BB * GWID * 4);
  float* latbuf = (float*)alloc((size_t)BB * 128 * 4);
  float* zbuf   = (float*)alloc((size_t)BB * 128 * 4);
  float* condb  = (float*)alloc((size_t)BB * 128 * 4);
  float* zcat   = (float*)alloc((size_t)BB * 256 * 4);
  float* zcbuf  = (float*)alloc((size_t)BB * 128 * 4);
  int* counts = (int*)alloc((size_t)NN * 4);
  int* starts = (int*)alloc((size_t)NN * 4);
  int* cursor = (int*)alloc((size_t)NN * 4);
  int* bsums  = (int*)alloc(256 * 4);
  int* slist  = (int*)alloc((size_t)(EE + NN) * 4);
  int* gstart = (int*)alloc((size_t)BB * 4);
  int* gcount = (int*)alloc((size_t)BB * 4);
  int* order  = (int*)alloc((size_t)NN * 4);

  const int NB = CDIV(NN, 256);   // 160

  // ---- CSR (dst-grouped, self loops included) + graph ranges ----
  k_fill_int<<<NB, 256, 0, stream>>>(counts, 1, NN);            // self loop
  k_fill_int<<<NB, 256, 0, stream>>>(cursor, 0, NN);
  k_fill_int<<<CDIV(BB,256), 256, 0, stream>>>(gstart, INT_MAX, BB);
  k_fill_int<<<CDIV(BB,256), 256, 0, stream>>>(gcount, 0, BB);
  k_count_dst<<<CDIV(EE,256), 256, 0, stream>>>(edge_index + EE, EE, counts);
  k_scan_block<<<NB, 256, 0, stream>>>(counts, starts, bsums, NN);
  k_scan_sums<<<1, 1, 0, stream>>>(bsums, NB);
  k_scan_add<<<NB, 256, 0, stream>>>(starts, bsums, NN);
  k_scatter_edges<<<CDIV(EE,256), 256, 0, stream>>>(edge_index, edge_index + EE, EE,
                                                    starts, cursor, slist);
  k_scatter_self<<<NB, 256, 0, stream>>>(NN, starts, cursor, slist);
  k_sort_lists<<<NB, 256, 0, stream>>>(starts, counts, slist, NN);
  k_graph_ranges<<<NB, 256, 0, stream>>>(batch, NN, gstart, gcount);
  k_order<<<NB, 256, 0, stream>>>(batch, gstart, order, NN);

  auto gemm = [&](const float* A, const float* Bw, float* C, const float* bias,
                  int M, int Ncols, int K, int relu) {
    dim3 grid(CDIV(M, BM), CDIV(Ncols, BN));
    k_gemm_bf16<<<grid, 256, 0, stream>>>(A, Bw, C, bias, M, Ncols, K, relu);
  };
  auto pool = [&](const float* x, int C, int colofs) {
    dim3 grid(BB, CDIV(C, 256));
    k_pool<<<grid, 256, 0, stream>>>(x, C, gstart, gcount, gbuf, colofs);
  };
  auto gat_layer = [&](const GatP& g, int K, const float* onehotW) {
    gemm(xbuf, g.W, xwbuf, nullptr, NN, HID, K, 0);               // xw = x @ W (WMMA)
    if (onehotW) k_add_onehot<<<NN, 256, 0, stream>>>(xwbuf, onehotW, order);
    k_att<<<CDIV(NN*NH,256), 256, 0, stream>>>(xwbuf, g.asrc, g.adst, a_src, a_dst);
    k_gat_aggregate<<<NN, 256, 0, stream>>>(xwbuf, a_src, a_dst, starts, counts,
                                            slist, g.b, xbuf);    // in-place ok
  };

  // ---- GraphEncoder ----
  k_enc_cat<<<CDIV(NN*256,256), 256, 0, stream>>>(node_features, enc_bbox_W, enc_bbox_b,
                                                  enc_mask_embed, building_mask, xwbuf);
  gemm(xwbuf, enc_node_W, xbuf, enc_node_b, NN, 128, 256, 1);     // n0
  pool(xbuf, 128, 0);
  for (int l = 0; l < 3; ++l) {
    gat_layer(enc_g[l], l == 0 ? 128 : HID, nullptr);
    pool(xbuf, HID, 128 + l * HID);
  }
  gemm(gbuf, enc_agg_W, latbuf, enc_agg_b, BB, 128, GWID, 0);     // lat
  gemm(latbuf, mu_W,  out + OUT_MU, mu_b,  BB, 128, 128, 0);      // mu -> d_out
  gemm(latbuf, var_W, out + OUT_LV, var_b, BB, 128, 128, 0);      // log_var -> d_out
  k_reparam<<<CDIV(BB*128,256), 256, 0, stream>>>(out + OUT_MU, out + OUT_LV, eps, zbuf);

  // ---- GraphConditionEncoder ----
  k_lin5_relu<<<CDIV(NN*128,256), 256, 0, stream>>>(cond_street, cond_bbox_W, cond_bbox_b, xbuf);
  pool(xbuf, 128, 0);
  for (int l = 0; l < 3; ++l) {
    gat_layer(cond_g[l], l == 0 ? 128 : HID, nullptr);
    pool(xbuf, HID, 128 + l * HID);
  }
  gemm(gbuf, cond_agg_W, condb, cond_agg_b, BB, 128, GWID, 0);    // condition

  // ---- GraphDecoder ----
  k_concat2<<<CDIV(BB*256,256), 256, 0, stream>>>(zbuf, condb, zcat);
  gemm(zcat, init_W, zcbuf, init_b, BB, 128, 256, 0);             // zc
  k_dec_input<<<CDIV(NN*128,256), 256, 0, stream>>>(zcbuf, batch, xbuf);
  // gat0: K=128 (zn part of W) + one-hot row gather for the 320 position columns
  gat_layer(dec_g[0], 128, dec_g[0].W);
  gat_layer(dec_g[1], HID, nullptr);
  gat_layer(dec_g[2], HID, nullptr);

  gemm(xbuf, pos_W, xwbuf, pos_b, NN, 128, HID, 1);
  k_head_out<<<CDIV(NN*2,256), 256, 0, stream>>>(xwbuf, fc_pos_W, fc_pos_b, out + OUT_POS, 2);
  gemm(xbuf, size_W, xwbuf, size_b, NN, 128, HID, 1);
  k_head_out<<<CDIV(NN*2,256), 256, 0, stream>>>(xwbuf, fc_size_W, fc_size_b, out + OUT_SIZE, 2);
  gemm(xbuf, theta_W, xwbuf, theta_b, NN, 128, HID, 1);
  k_head_out<<<CDIV(NN,256), 256, 0, stream>>>(xwbuf, fc_theta_W, fc_theta_b, out + OUT_THETA, 1);
}
